// HGA_33964601377204
// MI455X (gfx1250) — matrix operand order repaired and checked by reading, not compiled, not run on hardware
//
#include <hip/hip_runtime.h>

#define HB 8
#define HC 128
#define HN 4096
#define KSTR 136   // halfs per K-tile LDS row: 256B row + 16B pad
#define VSTR 72    // halfs per V-tile LDS row: 128B row + 16B pad
#define PSTR 72    // halfs per P LDS row (16 rows x 64 j per wave), padded

typedef __attribute__((ext_vector_type(16))) _Float16 v16h;
typedef __attribute__((ext_vector_type(8)))  _Float16 v8h;
typedef __attribute__((ext_vector_type(8)))  float    v8f;
typedef __attribute__((ext_vector_type(4)))  float    v4f;
typedef __attribute__((ext_vector_type(4)))  unsigned int u32x4;
typedef __attribute__((ext_vector_type(8)))  int      i32x8;
typedef __attribute__((ext_vector_type(4)))  int      i32x4;

#if defined(__has_builtin)
#if __has_builtin(__builtin_amdgcn_tensor_load_to_lds) && __has_builtin(__builtin_amdgcn_s_wait_tensorcnt)
#define USE_TDM 1
#endif
#endif

// 16-bit A fragment (16x32, MxK): per lane two contiguous 16B chunks at
// k = hi*8 and hi*8+16 (ISA 7.12.2 table). Element (m,k) = base[m*stride + k].
static __device__ __forceinline__ v16h frag_a(const _Float16* base, int stride, int lane) {
  int r = lane & 15, hi = (lane >> 4) & 1;
  const _Float16* p = base + r * stride + hi * 8;
  v8h a = *(const v8h*)(p);
  v8h b = *(const v8h*)(p + 16);
  return __builtin_shufflevector(a, b, 0,1,2,3,4,5,6,7,8,9,10,11,12,13,14,15);
}

// 16-bit B fragment (32x16, KxN): per lane two contiguous 16B chunks at
// k = hi*8 and hi*8+16 (same k order as frag_a).
// Element (k,n) = base[n*stride + k].
static __device__ __forceinline__ v16h frag_b(const _Float16* base, int stride, int lane) {
  int r = lane & 15, hi = (lane >> 4) & 1;
  const _Float16* p = base + r * stride + hi * 8;
  v8h a = *(const v8h*)(p);
  v8h b = *(const v8h*)(p + 16);
  return __builtin_shufflevector(a, b, 0,1,2,3,4,5,6,7,8,9,10,11,12,13,14,15);
}

#ifdef USE_TDM
// Issue a 2-D TDM tile load global->LDS. dims/tiles/strides in data_size=8B
// units; LDS row padding via D# pad fields. Uniform (scalar) operands only.
// 6-arg builtin form (clang-23 / therock-10.0 headers).
static __device__ __forceinline__ void tdm_load_2d(
    unsigned int lds_byte_off, const void* gptr, unsigned int d0_flags,
    unsigned int dim0, unsigned int dim1, unsigned int tile0,
    unsigned int tile1, unsigned int stride0)
{
  unsigned long long ga = (unsigned long long)(uintptr_t)gptr;
  u32x4 g0;
  g0[0] = 1u;                                     // count=1, user-mode load
  g0[1] = lds_byte_off;                           // lds_addr [63:32]
  g0[2] = (unsigned int)(ga & 0xffffffffu);       // global_addr lo
  g0[3] = (unsigned int)(ga >> 32) | 0x80000000u; // global_addr hi | type=2
  i32x8 g1;
  g1[0] = (int)d0_flags;                          // mask=0|data_size|pad cfg
  g1[1] = (int)((dim0 & 0xffffu) << 16);          // atomic_addr=0 | dim0[15:0]
  g1[2] = (int)((dim0 >> 16) | ((dim1 & 0xffffu) << 16));
  g1[3] = (int)((dim1 >> 16) | (tile0 << 16));    // dim1[31:16] | tile_dim0
  g1[4] = (int)(tile1 & 0xffffu);                 // tile_dim1 | tile_dim2=0
  g1[5] = (int)stride0;                           // dim0_stride[31:0]
  g1[6] = 0;                                      // stride hi bits
  g1[7] = 0;
  i32x4 z4 = {0, 0, 0, 0};
  i32x8 z8 = {0, 0, 0, 0, 0, 0, 0, 0};
  __builtin_amdgcn_tensor_load_to_lds(g0, g1, z4, z4, z8, 0);
}
// data_size=8B (3<<16) | pad_enable (1<<20) | pad_interval<<22 | pad_amount<<25
#define TDM_FLAGS_K ((3u << 16) | (1u << 20) | (5u << 22) | (3u << 25)) // 64 DW rows, +4 DW pad
#define TDM_FLAGS_V ((3u << 16) | (1u << 20) | (4u << 22) | (3u << 25)) // 32 DW rows, +4 DW pad
#endif

// ---------------------------------------------------------------------------
// Projections: q/k -> [B,N,C] f16, v -> [B,C,N] f16. f32 math, W cached in LDS.
// ---------------------------------------------------------------------------
__global__ __launch_bounds__(256) void hga_proj(
    const float* __restrict__ x1, const float* __restrict__ x2,
    const float* __restrict__ Wq, const float* __restrict__ bq,
    const float* __restrict__ Wk, const float* __restrict__ bk,
    const float* __restrict__ Wv, const float* __restrict__ bv,
    _Float16* __restrict__ Qg, _Float16* __restrict__ Kg, _Float16* __restrict__ Vg)
{
  __shared__ __align__(16) float Wl[HC * HC];
  const int z = blockIdx.z;
  const float* X    = (z == 0) ? x1 : x2;
  const float* W    = (z == 0) ? Wq : (z == 1 ? Wk : Wv);
  const float* bias = (z == 0) ? bq : (z == 1 ? bk : bv);
  const int b   = blockIdx.y;
  const int tid = threadIdx.x;

  for (int i = tid; i < HC * HC / 4; i += 256)
    ((v4f*)Wl)[i] = ((const v4f*)W)[i];
  __syncthreads();

  const int n = blockIdx.x * 256 + tid;
  const float* xp = X + (size_t)b * HC * HN + n;

  for (int oc = 0; oc < 8; ++oc) {
    float acc[16];
    #pragma unroll
    for (int i = 0; i < 16; ++i) acc[i] = bias[oc * 16 + i];
    for (int c0 = 0; c0 < HC; c0 += 4) {
      float xv[4];
      #pragma unroll
      for (int cc = 0; cc < 4; ++cc) xv[cc] = xp[(size_t)(c0 + cc) * HN];
      #pragma unroll
      for (int i = 0; i < 16; ++i) {
        v4f w = *(const v4f*)(Wl + (oc * 16 + i) * HC + c0);
        acc[i] = fmaf(w[0], xv[0], acc[i]);
        acc[i] = fmaf(w[1], xv[1], acc[i]);
        acc[i] = fmaf(w[2], xv[2], acc[i]);
        acc[i] = fmaf(w[3], xv[3], acc[i]);
      }
    }
    if (z == 0) {
      #pragma unroll
      for (int i = 0; i < 16; ++i)
        Qg[((size_t)b * HN + n) * HC + oc * 16 + i] = (_Float16)acc[i];
    } else if (z == 1) {
      #pragma unroll
      for (int i = 0; i < 16; ++i)
        Kg[((size_t)b * HN + n) * HC + oc * 16 + i] = (_Float16)acc[i];
    } else {
      #pragma unroll
      for (int i = 0; i < 16; ++i)
        Vg[((size_t)b * HC + oc * 16 + i) * HN + n] = (_Float16)acc[i];
    }
  }
}

// ---------------------------------------------------------------------------
// Fused flash attention, double-buffered K/V tiles via TDM (tensor_load_to_lds)
// issued by waves 0/1 while all 4 waves run the WMMA chain on the other buffer.
// ---------------------------------------------------------------------------
__global__ __launch_bounds__(128) void hga_attn(
    const _Float16* __restrict__ Qg, const _Float16* __restrict__ Kg,
    const _Float16* __restrict__ Vg, float* __restrict__ out)
{
  __shared__ __align__(16) _Float16 lds_k[2][64 * KSTR];
  __shared__ __align__(16) _Float16 lds_v[2][HC * VSTR];
  __shared__ __align__(16) _Float16 lds_p[4 * 16 * PSTR];

  const int tid  = threadIdx.x;
  const int lane = tid & 31;
  const int wave = tid >> 5;
  const int b    = blockIdx.y;
  const int i0   = blockIdx.x * 64;
  const int r15  = lane & 15;
  const int hi   = (lane >> 4) & 1;

  // Q rows for this wave: 4 A-fragments over k = 0..127, straight from global.
  const _Float16* qbase = Qg + ((size_t)b * HN + i0 + wave * 16) * HC;
  v16h qf[4];
  #pragma unroll
  for (int kk = 0; kk < 4; ++kk) qf[kk] = frag_a(qbase + kk * 32, HC, lane);

  v8f o[8];
  #pragma unroll
  for (int ct = 0; ct < 8; ++ct) o[ct] = (v8f){};
  float mr[8], lr[8];
  #pragma unroll
  for (int r = 0; r < 8; ++r) { mr[r] = -1e30f; lr[r] = 0.0f; }

  _Float16* pw = lds_p + wave * 16 * PSTR;

#ifdef USE_TDM
  // prologue: DMA tile 0 into buffer 0 (wave0: K, wave1: V)
  if (wave == 0)
    tdm_load_2d((unsigned int)(uintptr_t)(void*)&lds_k[0][0],
                Kg + ((size_t)b * HN + 0) * HC, TDM_FLAGS_K,
                /*dim0*/32, /*dim1*/HN, /*tile0*/32, /*tile1*/64, /*stride0*/32);
  else if (wave == 1)
    tdm_load_2d((unsigned int)(uintptr_t)(void*)&lds_v[0][0],
                Vg + (size_t)b * HC * HN + 0, TDM_FLAGS_V,
                /*dim0*/HN / 4, /*dim1*/HC, /*tile0*/16, /*tile1*/HC, /*stride0*/HN / 4);
#endif

  for (int t = 0; t < HN / 64; ++t) {
    const int cur = t & 1;
    const int j0  = t * 64;

#ifdef USE_TDM
    if (wave < 2) __builtin_amdgcn_s_wait_tensorcnt(0);
    __syncthreads();  // buf[cur] DMA complete; buf[cur^1] no longer read
    if (t + 1 < HN / 64) {
      const int jn = j0 + 64;
      if (wave == 0)
        tdm_load_2d((unsigned int)(uintptr_t)(void*)&lds_k[cur ^ 1][0],
                    Kg + ((size_t)b * HN + jn) * HC, TDM_FLAGS_K,
                    32, HN, 32, 64, 32);
      else if (wave == 1)
        tdm_load_2d((unsigned int)(uintptr_t)(void*)&lds_v[cur ^ 1][0],
                    Vg + (size_t)b * HC * HN + jn, TDM_FLAGS_V,
                    HN / 4, HC, 16, HC, HN / 4);
    }
#else
    __syncthreads();  // previous tiles fully consumed
    for (int idx = tid; idx < 64 * 16; idx += 128) {
      int row = idx >> 4, ch = idx & 15;
      *(v8h*)(&lds_k[cur][0] + row * KSTR + ch * 8) =
          *(const v8h*)(Kg + ((size_t)b * HN + j0 + row) * HC + ch * 8);
    }
    for (int idx = tid; idx < 128 * 8; idx += 128) {
      int row = idx >> 3, ch = idx & 7;
      *(v8h*)(&lds_v[cur][0] + row * VSTR + ch * 8) =
          *(const v8h*)(Vg + ((size_t)b * HC + row) * HN + j0 + ch * 8);
    }
    __syncthreads();
#endif

    const _Float16* kb = &lds_k[cur][0];
    const _Float16* vb = &lds_v[cur][0];

    // S[16x64] = Q * K_tile^T ; B-frag element (k=c, n=j) = lds_k[j][c]
    v8f s[4];
    #pragma unroll
    for (int jt = 0; jt < 4; ++jt) {
      v8f acc = (v8f){};
      #pragma unroll
      for (int kk = 0; kk < 4; ++kk) {
        v16h kf = frag_b(kb + jt * 16 * KSTR + kk * 32, KSTR, lane);
        acc = __builtin_amdgcn_wmma_f32_16x16x32_f16(
            false, qf[kk], false, kf, (short)0, acc, false, false);
      }
      s[jt] = acc;
    }

    // ---- online softmax (row = VGPR idx r + 8*hi; 16 lanes/half share a row)
    float nm[8];
    #pragma unroll
    for (int r = 0; r < 8; ++r) nm[r] = mr[r];
    #pragma unroll
    for (int jt = 0; jt < 4; ++jt)
      #pragma unroll
      for (int r = 0; r < 8; ++r) nm[r] = fmaxf(nm[r], s[jt][r]);
    #pragma unroll
    for (int off = 8; off >= 1; off >>= 1)
      #pragma unroll
      for (int r = 0; r < 8; ++r) nm[r] = fmaxf(nm[r], __shfl_xor(nm[r], off, 32));

    float sc[8];
    #pragma unroll
    for (int r = 0; r < 8; ++r) { sc[r] = __expf(mr[r] - nm[r]); mr[r] = nm[r]; }
    #pragma unroll
    for (int ct = 0; ct < 8; ++ct)
      #pragma unroll
      for (int r = 0; r < 8; ++r) o[ct][r] *= sc[r];

    float ps[8];
    #pragma unroll
    for (int r = 0; r < 8; ++r) ps[r] = 0.0f;
    #pragma unroll
    for (int jt = 0; jt < 4; ++jt)
      #pragma unroll
      for (int r = 0; r < 8; ++r) {
        float p = __expf(s[jt][r] - nm[r]);
        s[jt][r] = p;
        ps[r] += p;
      }
    #pragma unroll
    for (int off = 8; off >= 1; off >>= 1)
      #pragma unroll
      for (int r = 0; r < 8; ++r) ps[r] += __shfl_xor(ps[r], off, 32);
    #pragma unroll
    for (int r = 0; r < 8; ++r) lr[r] = lr[r] * sc[r] + ps[r];

    // P (C-layout) -> per-wave LDS scratch as f16 (re-shape to A-layout)
    #pragma unroll
    for (int jt = 0; jt < 4; ++jt)
      #pragma unroll
      for (int r = 0; r < 8; ++r)
        pw[(r + 8 * hi) * PSTR + jt * 16 + r15] = (_Float16)s[jt][r];

    // O += P * V_tile ; A = P (two k32 frags), B element (k=j, n=c) = lds_v[c][j]
    v16h pf[2];
    #pragma unroll
    for (int h = 0; h < 2; ++h) pf[h] = frag_a(pw + h * 32, PSTR, lane);
    #pragma unroll
    for (int ct = 0; ct < 8; ++ct) {
      #pragma unroll
      for (int h = 0; h < 2; ++h) {
        v16h vf = frag_b(vb + ct * 16 * VSTR + h * 32, VSTR, lane);
        o[ct] = __builtin_amdgcn_wmma_f32_16x16x32_f16(
            false, pf[h], false, vf, (short)0, o[ct], false, false);
      }
    }
  }

  // normalize and write out[b][c][i] (f32)
  float inv[8];
  #pragma unroll
  for (int r = 0; r < 8; ++r) inv[r] = 1.0f / lr[r];
  const int i_base = i0 + wave * 16 + 8 * hi;
  #pragma unroll
  for (int ct = 0; ct < 8; ++ct) {
    int c = ct * 16 + r15;
    #pragma unroll
    for (int r = 0; r < 8; ++r)
      out[((size_t)b * HC + c) * HN + (i_base + r)] = o[ct][r] * inv[r];
  }
}

extern "C" void kernel_launch(void* const* d_in, const int* in_sizes, int n_in,
                              void* d_out, int out_size, void* d_ws, size_t ws_size,
                              hipStream_t stream) {
  (void)in_sizes; (void)n_in; (void)out_size; (void)ws_size;
  const float* x1 = (const float*)d_in[0];
  const float* x2 = (const float*)d_in[1];
  const float* Wq = (const float*)d_in[2];
  const float* bq = (const float*)d_in[3];
  const float* Wk = (const float*)d_in[4];
  const float* bk = (const float*)d_in[5];
  const float* Wv = (const float*)d_in[6];
  const float* bv = (const float*)d_in[7];
  float* out = (float*)d_out;

  _Float16* Qg = (_Float16*)d_ws;                      // [B,N,C] f16, 8 MB
  _Float16* Kg = Qg + (size_t)HB * HN * HC;            // [B,N,C] f16, 8 MB
  _Float16* Vg = Kg + (size_t)HB * HN * HC;            // [B,C,N] f16, 8 MB

  hga_proj<<<dim3(HN / 256, HB, 3), 256, 0, stream>>>(
      x1, x2, Wq, bq, Wk, bk, Wv, bv, Qg, Kg, Vg);
  hga_attn<<<dim3(HN / 64, HB), 128, 0, stream>>>(Qg, Kg, Vg, out);
}
